// CausalSelfAttention_63471026700614
// MI455X (gfx1250) — matrix operand order repaired and checked by reading, not compiled, not run on hardware
//
#include <hip/hip_runtime.h>

#define DEV __device__ __forceinline__

typedef __attribute__((ext_vector_type(16))) __bf16 v16bf;
typedef __attribute__((ext_vector_type(8)))  float  v8f;
typedef __attribute__((ext_vector_type(4)))  int    v4i;

union FragBF {
  uint4 u[2];
  v16bf b;
};

DEV unsigned short f32_to_bf16(float f) {
  union { float f; unsigned u; } v; v.f = f;
  unsigned r = v.u + 0x7FFFu + ((v.u >> 16) & 1u);   // round-to-nearest-even
  return (unsigned short)(r >> 16);
}

// ---------------------------------------------------------------------------
// Async global->LDS 16-byte copy (CDNA5 GLOBAL_LOAD_ASYNC_TO_LDS_B128,
// tracked by ASYNCcnt), with a synchronous fallback if the builtin is absent.
// ---------------------------------------------------------------------------
#if defined(__has_builtin)
#if __has_builtin(__builtin_amdgcn_global_load_async_to_lds_b128)
#define HAVE_ASYNC_LDS 1
#endif
#endif
#ifndef HAVE_ASYNC_LDS
#define HAVE_ASYNC_LDS 0
#endif

typedef __attribute__((address_space(1))) v4i* gas1_v4ip;
typedef __attribute__((address_space(3))) v4i* as3_v4ip;

DEV void async_ld16(unsigned short* ldsp, const unsigned short* gp) {
#if HAVE_ASYNC_LDS
  __builtin_amdgcn_global_load_async_to_lds_b128(
      (gas1_v4ip)(unsigned long long)gp,
      (as3_v4ip)(unsigned)(unsigned long long)ldsp, 0, 0);
#else
  *(uint4*)ldsp = *(const uint4*)gp;
#endif
}

DEV void wait_async_lds() {
  asm volatile("s_wait_asynccnt 0" ::: "memory");
}

// ---------------------------------------------------------------------------
// fp32 -> bf16 conversion, 4 elements per thread
// ---------------------------------------------------------------------------
__global__ __launch_bounds__(256)
void cvt_f32_bf16x4(const float* __restrict__ in, unsigned short* __restrict__ out, int n4) {
  int i = blockIdx.x * blockDim.x + threadIdx.x;
  if (i < n4) {
    float4 f = ((const float4*)in)[i];
    union { unsigned long long u; unsigned short s[4]; } o;
    o.s[0] = f32_to_bf16(f.x);
    o.s[1] = f32_to_bf16(f.y);
    o.s[2] = f32_to_bf16(f.z);
    o.s[3] = f32_to_bf16(f.w);
    ((unsigned long long*)out)[i] = o.u;
  }
}

// ---------------------------------------------------------------------------
// bf16 GEMM, 128x128 block tile, 8 waves (2x4), 64x32 per wave, K-step 32,
// double-buffered LDS with async staging, one barrier per K-step.
// MODE 0: scatter into q*scale [B,H,T,D], kT [B,H,D,T], v [B,H,T,D] (N = 3C).
// MODE 1: fp32 row-major output.
// ---------------------------------------------------------------------------
template<int MODE>
__global__ __launch_bounds__(256)
void gemm_bf16_128x128(const unsigned short* __restrict__ A,
                       const unsigned short* __restrict__ Bm,
                       int M, int N, int K,
                       float* __restrict__ outF,
                       unsigned short* __restrict__ qOut,
                       unsigned short* __restrict__ kOut,
                       unsigned short* __restrict__ vOut)
{
  constexpr int LDA = 40;    // 128 x 32 tile, padded; 80B rows, 16B-aligned
  constexpr int LDB = 136;   // 32 x 128 tile, padded; 272B rows, 16B-aligned
  __shared__ unsigned short ldsA[2][128 * LDA];
  __shared__ unsigned short ldsB[2][32 * LDB];

  const int tid   = threadIdx.x;
  const int lane  = tid & 31;
  const int wid   = tid >> 5;
  const int half  = lane >> 4;
  const int l16   = lane & 15;
  const int waveM = wid & 1;    // 2 waves along M (64 rows each)
  const int waveN = wid >> 1;   // 4 waves along N (32 cols each)

  const int m0 = blockIdx.y * 128;
  const int n0 = blockIdx.x * 128;

  v8f zero = {};
  v8f acc[4][2];
  for (int mt = 0; mt < 4; ++mt)
    for (int nt = 0; nt < 2; ++nt) acc[mt][nt] = zero;

  // cooperative staging coordinates
  const int ar = tid >> 1, ac = (tid & 1) * 16;   // A: 128 rows x 32 cols
  const int br = tid >> 3, bc = (tid & 7) * 16;   // B: 32 rows x 128 cols

  auto stage = [&](int buf, int k0) {
    async_ld16(&ldsA[buf][ar * LDA + ac],     A + (size_t)(m0 + ar) * K + k0 + ac);
    async_ld16(&ldsA[buf][ar * LDA + ac + 8], A + (size_t)(m0 + ar) * K + k0 + ac + 8);
    async_ld16(&ldsB[buf][br * LDB + bc],     Bm + (size_t)(k0 + br) * N + n0 + bc);
    async_ld16(&ldsB[buf][br * LDB + bc + 8], Bm + (size_t)(k0 + br) * N + n0 + bc + 8);
  };

  const int nk = K >> 5;
  stage(0, 0);
  for (int kk = 0; kk < nk; ++kk) {
    wait_async_lds();
    __syncthreads();   // buf[kk&1] staged & visible; waves can't drift >1 iter,
                       // so staging buf[(kk+1)&1] below can't collide.
    if (kk + 1 < nk) stage((kk + 1) & 1, (kk + 1) * 32);

    const unsigned short* la = ldsA[kk & 1];
    const unsigned short* lb = ldsB[kk & 1];

    // B fragments (32x16): lane holds column N=l16, K chunks {half*8, 16+half*8}.
    FragBF fb[2];
    for (int nt = 0; nt < 2; ++nt) {
      const int cb = waveN * 32 + nt * 16;
      for (int s = 0; s < 16; ++s)
        ((unsigned short*)&fb[nt])[s] = lb[((s & 8) * 2 + half * 8 + (s & 7)) * LDB + cb + l16];
    }
    for (int mt = 0; mt < 4; ++mt) {
      // A fragment (16x32): lane holds row M=l16, K chunks {half*8, 16+half*8}.
      FragBF fa;
      const int row = waveM * 64 + mt * 16 + l16;
      fa.u[0] = *(const uint4*)&la[row * LDA + half * 8];
      fa.u[1] = *(const uint4*)&la[row * LDA + 16 + half * 8];
      for (int nt = 0; nt < 2; ++nt)
        acc[mt][nt] = __builtin_amdgcn_wmma_f32_16x16x32_bf16(
            false, fa.b, false, fb[nt].b, (short)0, acc[mt][nt], false, false);
    }
  }

  // Epilogue. C layout: lane holds column n=l16, rows M = half*8 + r.
  for (int mt = 0; mt < 4; ++mt)
    for (int nt = 0; nt < 2; ++nt)
      for (int r = 0; r < 8; ++r) {
        const int m = m0 + waveM * 64 + mt * 16 + half * 8 + r;
        const int n = n0 + waveN * 32 + nt * 16 + l16;
        float val = acc[mt][nt][r];
        if (MODE == 0) {
          const int b = m >> 11, t = m & 2047;          // T = 2048
          const int which = n >> 10, c = n & 1023;      // C = 1024
          const int h = c >> 6, d = c & 63;             // D = 64
          if (which == 1) {         // K stored transposed: [B,H,D,T]
            kOut[(((size_t)(b * 16 + h) * 64 + d) << 11) + t] = f32_to_bf16(val);
          } else {
            if (which == 0) val *= 0.18033688f;   // fold (1/sqrt(D))*log2(e) into q
            unsigned short* dst = (which == 0) ? qOut : vOut;
            dst[((size_t)((b * 16 + h) * 2048 + t) << 6) + d] = f32_to_bf16(val);
          }
        } else {
          outF[(size_t)m * N + n] = val;
        }
      }
}

// ---------------------------------------------------------------------------
// Flash attention: 128 queries per block (8 waves x 16 rows), key tiles of 64,
// double-buffered async K/V staging (one barrier per tile), online softmax in
// exp2 domain (scale pre-folded into q), causal. q/v bf16 [B,H,T,D], kT bf16
// [B,H,D,T]; output y bf16 [B,T,C].
// ---------------------------------------------------------------------------
__global__ __launch_bounds__(256)
void flash_attn(const unsigned short* __restrict__ Q,
                const unsigned short* __restrict__ Kt,
                const unsigned short* __restrict__ Vm,
                unsigned short* __restrict__ Y)
{
  constexpr int LDS_T = 72;   // padded stride (144B rows, 16B-aligned)
  __shared__ unsigned short ldsK[2][64 * LDS_T];   // [d][key]  (K^T tile)
  __shared__ unsigned short ldsV[2][64 * LDS_T];   // [key][d]
  __shared__ unsigned short ldsP[8][16 * LDS_T];   // per-wave P staging

  const int tid  = threadIdx.x;
  const int lane = tid & 31;
  const int wid  = tid >> 5;
  const int half = lane >> 4;
  const int l16  = lane & 15;
  const int bh   = blockIdx.y;
  const int q0   = blockIdx.x * 128;
  const int qr   = q0 + wid * 16;   // this wave's first query row

  const unsigned short* Qb  = Q  + (size_t)bh * 2048 * 64;
  const unsigned short* Ktb = Kt + (size_t)bh * 64 * 2048;
  const unsigned short* Vb  = Vm + (size_t)bh * 2048 * 64;

  // Q fragments held in registers for the whole block (D=64 -> 2 K-chunks).
  FragBF qf[2];
  {
    const int row = qr + l16;
    for (int kc = 0; kc < 2; ++kc) {
      qf[kc].u[0] = *(const uint4*)(Qb + (size_t)row * 64 + kc * 32 + half * 8);
      qf[kc].u[1] = *(const uint4*)(Qb + (size_t)row * 64 + kc * 32 + 16 + half * 8);
    }
  }

  float mrow[8], lrow[8];
  v8f zero = {};
  v8f oacc[4];
  for (int r = 0; r < 8; ++r) { mrow[r] = -3.0e38f; lrow[r] = 0.f; }
  for (int nt = 0; nt < 4; ++nt) oacc[nt] = zero;

  const int nkt  = q0 / 64 + 2;        // causal: key tiles up to the diagonal
  const int srow = tid >> 2;           // staging: 64 rows, 16 elems/thread
  const int sc0  = (tid & 3) * 16;

  auto stageKV = [&](int buf, int kt) {
    async_ld16(&ldsK[buf][srow * LDS_T + sc0],
               Ktb + ((size_t)srow << 11) + kt * 64 + sc0);
    async_ld16(&ldsK[buf][srow * LDS_T + sc0 + 8],
               Ktb + ((size_t)srow << 11) + kt * 64 + sc0 + 8);
    async_ld16(&ldsV[buf][srow * LDS_T + sc0],
               Vb + (size_t)(kt * 64 + srow) * 64 + sc0);
    async_ld16(&ldsV[buf][srow * LDS_T + sc0 + 8],
               Vb + (size_t)(kt * 64 + srow) * 64 + sc0 + 8);
  };

  stageKV(0, 0);
  for (int kt = 0; kt < nkt; ++kt) {
    wait_async_lds();
    __syncthreads();   // buf[kt&1] ready; single barrier per tile (see GEMM note)
    if (kt + 1 < nkt) stageKV((kt + 1) & 1, kt + 1);

    const unsigned short* lk = ldsK[kt & 1];
    const unsigned short* lv = ldsV[kt & 1];

    // Waves entirely above the diagonal skip all compute for this tile.
    if (kt * 64 <= qr + 15) {
      // S = Q * K^T  (16 queries x 64 keys per wave); scale pre-folded into Q.
      v8f sacc[4];
      for (int nt = 0; nt < 4; ++nt) sacc[nt] = zero;
      for (int kc = 0; kc < 2; ++kc)
        for (int nt = 0; nt < 4; ++nt) {
          FragBF fb;   // B: lane holds key column nt*16+l16, d chunks {half*8, 16+half*8}
          for (int s = 0; s < 16; ++s)
            ((unsigned short*)&fb)[s] = lk[(kc * 32 + (s & 8) * 2 + half * 8 + (s & 7)) * LDS_T + nt * 16 + l16];
          sacc[nt] = __builtin_amdgcn_wmma_f32_16x16x32_bf16(
              false, qf[kc].b, false, fb.b, (short)0, sacc[nt], false, false);
        }

      // Causal mask only on diagonal tiles.
      if (kt * 64 + 63 > qr) {
        for (int nt = 0; nt < 4; ++nt) {
          const int key = kt * 64 + nt * 16 + l16;
          for (int r = 0; r < 8; ++r) {
            const int qrow = qr + half * 8 + r;
            if (key > qrow) sacc[nt][r] = -3.0e38f;
          }
        }
      }

      // Online softmax (exp2 domain); rows live across one 16-lane half.
      for (int r = 0; r < 8; ++r) {
        float t = fmaxf(fmaxf(sacc[0][r], sacc[1][r]), fmaxf(sacc[2][r], sacc[3][r]));
        for (int off = 1; off < 16; off <<= 1)
          t = fmaxf(t, __shfl_xor(t, off, 32));
        const float mnew = fmaxf(mrow[r], t);
        const float corr = exp2f(mrow[r] - mnew);
        mrow[r] = mnew;
        lrow[r] *= corr;
        for (int nt = 0; nt < 4; ++nt) oacc[nt][r] *= corr;
        float rs = 0.f;
        for (int nt = 0; nt < 4; ++nt) {
          const float p = exp2f(sacc[nt][r] - mnew);
          sacc[nt][r] = p;
          rs += p;
        }
        for (int off = 1; off < 16; off <<= 1)
          rs += __shfl_xor(rs, off, 32);
        lrow[r] += rs;
      }

      // C-layout -> A-layout for P via per-wave LDS pad
      unsigned short* pw = &ldsP[wid][0];
      for (int nt = 0; nt < 4; ++nt)
        for (int r = 0; r < 8; ++r)
          pw[(half * 8 + r) * LDS_T + nt * 16 + l16] = f32_to_bf16(sacc[nt][r]);
      asm volatile("s_wait_dscnt 0" ::: "memory");   // intra-wave DS RAW

      // O += P * V
      for (int kc = 0; kc < 2; ++kc) {
        FragBF pa;
        pa.u[0] = *(const uint4*)&pw[l16 * LDS_T + kc * 32 + half * 8];
        pa.u[1] = *(const uint4*)&pw[l16 * LDS_T + kc * 32 + 16 + half * 8];
        for (int nt = 0; nt < 4; ++nt) {
          FragBF fb;   // B = V: lane holds d column nt*16+l16, key chunks {half*8, 16+half*8}
          for (int s = 0; s < 16; ++s)
            ((unsigned short*)&fb)[s] = lv[(kc * 32 + (s & 8) * 2 + half * 8 + (s & 7)) * LDS_T + nt * 16 + l16];
          oacc[nt] = __builtin_amdgcn_wmma_f32_16x16x32_bf16(
              false, pa.b, false, fb.b, (short)0, oacc[nt], false, false);
        }
      }
    }
  }

  // y[b, t, h*64 + d] = O / l   (bf16); one reciprocal per row
  const int b = bh >> 4, h = bh & 15;
  float rinv[8];
  for (int r = 0; r < 8; ++r) rinv[r] = 1.0f / lrow[r];
  for (int nt = 0; nt < 4; ++nt)
    for (int r = 0; r < 8; ++r) {
      const int t = qr + half * 8 + r;
      const int col = h * 64 + nt * 16 + l16;
      const float v = oacc[nt][r] * rinv[r];
      Y[((size_t)(b * 2048 + t) << 10) + col] = f32_to_bf16(v);
    }
}

// ---------------------------------------------------------------------------
extern "C" void kernel_launch(void* const* d_in, const int* in_sizes, int n_in,
                              void* d_out, int out_size, void* d_ws, size_t ws_size,
                              hipStream_t stream)
{
  const float* x     = (const float*)d_in[0];   // [4, 2048, 1024]
  const float* Wqkv  = (const float*)d_in[1];   // [1024, 3072]
  const float* Wproj = (const float*)d_in[2];   // [1024, 1024]
  float* out = (float*)d_out;                   // [4, 2048, 1024] fp32

  char* ws = (char*)d_ws;
  size_t off = 0;
  auto alloc = [&](size_t bytes) -> void* {
    void* p = ws + off;
    off += (bytes + 255) & ~(size_t)255;
    return p;
  };
  unsigned short* xb     = (unsigned short*)alloc((size_t)8192 * 1024 * 2);
  unsigned short* wqkvb  = (unsigned short*)alloc((size_t)1024 * 3072 * 2);
  unsigned short* wprojb = (unsigned short*)alloc((size_t)1024 * 1024 * 2);
  unsigned short* qb     = (unsigned short*)alloc((size_t)64 * 2048 * 64 * 2);
  unsigned short* kb     = (unsigned short*)alloc((size_t)64 * 2048 * 64 * 2);  // [B,H,D,T]
  unsigned short* vb     = (unsigned short*)alloc((size_t)64 * 2048 * 64 * 2);
  unsigned short* yb     = (unsigned short*)alloc((size_t)8192 * 1024 * 2);

  // fp32 -> bf16
  cvt_f32_bf16x4<<<(8192 * 1024 / 4) / 256, 256, 0, stream>>>(x, xb, 8192 * 1024 / 4);
  cvt_f32_bf16x4<<<(1024 * 3072 / 4) / 256, 256, 0, stream>>>(Wqkv, wqkvb, 1024 * 3072 / 4);
  cvt_f32_bf16x4<<<(1024 * 1024 / 4) / 256, 256, 0, stream>>>(Wproj, wprojb, 1024 * 1024 / 4);

  // QKV projection: [8192,1024] x [1024,3072] -> q*scale, kT [B,H,D,T], v
  gemm_bf16_128x128<0><<<dim3(3072 / 128, 8192 / 128), 256, 0, stream>>>(
      xb, wqkvb, 8192, 3072, 1024, nullptr, qb, kb, vb);

  // Causal flash attention -> y [B,T,C] bf16
  flash_attn<<<dim3(2048 / 128, 64), 256, 0, stream>>>(qb, kb, vb, yb);

  // Output projection: [8192,1024] x [1024,1024] -> fp32 out
  gemm_bf16_128x128<1><<<dim3(1024 / 128, 8192 / 128), 256, 0, stream>>>(
      yb, wprojb, 8192, 1024, 1024, out, nullptr, nullptr, nullptr);
}
